// GNN_att_71588514890560
// MI455X (gfx1250) — matrix operand order repaired and checked by reading, not compiled, not run on hardware
//
#include <hip/hip_runtime.h>

// ---------------------------------------------------------------------------
// CDNA5 WMMA types (wave32): A/B = 16 bf16 (8 VGPRs), C/D = 8 f32
// ---------------------------------------------------------------------------
typedef __attribute__((ext_vector_type(16))) __bf16 v16bf;
typedef __attribute__((ext_vector_type(8)))  __bf16 v8bf;
typedef __attribute__((ext_vector_type(8)))  float  v8f;

__device__ inline __bf16 f2bf(float f) {
    unsigned u = __builtin_bit_cast(unsigned, f);
    u += 0x7FFFu + ((u >> 16) & 1u);          // round-to-nearest-even
    unsigned short h = (unsigned short)(u >> 16);
    return __builtin_bit_cast(__bf16, h);
}

// A k order used here (contiguous, same as B loads): lanes 0-15 K in {0..15}, lanes 16-31 K in {16..31}
__device__ inline int a_kbase(int g, int half) {
    return 2 * g + half * 16;
}

__device__ inline v16bf cat8(v8bf lo, v8bf hi) {
    return __builtin_shufflevector(lo, hi, 0, 1, 2, 3, 4, 5, 6, 7,
                                           8, 9, 10, 11, 12, 13, 14, 15);
}

__device__ inline float siluf(float x) { return x / (1.f + __expf(-x)); }

// ---------------------------------------------------------------------------
// utility kernels
// ---------------------------------------------------------------------------
__global__ void k_zero(float* __restrict__ p, long n) {
    long i = (long)blockIdx.x * blockDim.x + threadIdx.x;
    if (i < n) p[i] = 0.f;
}

// edge scatter-sum: out[dst] += X[src], D features per node
__global__ void k_scatter(const float* __restrict__ X, const int* __restrict__ ei,
                          long E, int D, float* __restrict__ out) {
    long i = (long)blockIdx.x * blockDim.x + threadIdx.x;
    if (i >= E * (long)D) return;
    long e = i / D;
    int  j = (int)(i - e * (long)D);
    int  s = ei[e];
    int  d = ei[E + e];
    atomicAdd(&out[(size_t)d * D + j], X[(size_t)s * D + j]);
}

// out[m*Kp+k] = bf16(A[m*K+k] + B[m*K+k]) for k<K else 0   (zero-padded fuse)
__global__ void k_fuse_bf16(const float* __restrict__ A, const float* __restrict__ B,
                            __bf16* __restrict__ out, long M, int K, int Kp) {
    long i = (long)blockIdx.x * blockDim.x + threadIdx.x;
    if (i >= M * (long)Kp) return;
    long m = i / Kp;
    int  k = (int)(i - m * (long)Kp);
    out[i] = (k < K) ? f2bf(A[m * K + k] + B[m * K + k]) : f2bf(0.f);
}

// out[n*Kp+k] = bf16(W[k*N+n]) for k<K else 0   (weight transpose + pad)
__global__ void k_wT_bf16(const float* __restrict__ W, __bf16* __restrict__ out,
                          int K, int N, int Kp) {
    int i = blockIdx.x * blockDim.x + threadIdx.x;
    if (i >= N * Kp) return;
    int n = i / Kp, k = i % Kp;
    out[i] = (k < K) ? f2bf(W[(size_t)k * N + n]) : f2bf(0.f);
}

// ---------------------------------------------------------------------------
// C[M,Nn](f32) = Ab[M,Kp](bf16) @ Bt[Nn,Kp](bf16)^T + bias
// one wave per 16x16 tile; fragments are pure vector loads.
// ---------------------------------------------------------------------------
__global__ __launch_bounds__(128)
void k_gemm_bf16(const __bf16* __restrict__ Ab, const __bf16* __restrict__ Bt,
                 const float* __restrict__ bias, float* __restrict__ C,
                 int M, int Nn, int Kp) {
    const int wid    = blockIdx.x * (blockDim.x >> 5) + (threadIdx.x >> 5);
    const int ntiles = Nn >> 4;
    const int mt = wid / ntiles, nt = wid % ntiles;
    if (mt * 16 >= M) return;
    const int lane = threadIdx.x & 31, half = lane >> 4, l = lane & 15;
    const __bf16* Ap = Ab + (size_t)(mt * 16 + l) * Kp + half * 16;
    const __bf16* Bp = Bt + (size_t)(nt * 16 + l) * Kp + half * 16;

    v8f acc = {};
    for (int kc = 0; kc < Kp; kc += 32) {
        v8bf  alo = *(const v8bf*)(Ap + kc);
        v8bf  ahi = *(const v8bf*)(Ap + kc + 8);
        v16bf a   = cat8(alo, ahi);
        v16bf b   = *(const v16bf*)(Bp + kc);
        acc = __builtin_amdgcn_wmma_f32_16x16x32_bf16(false, a, false, b,
                                                      (short)0, acc, false, false);
    }
    const float bv = bias[nt * 16 + l];
#pragma unroll
    for (int r = 0; r < 8; ++r)
        C[(size_t)(mt * 16 + r + half * 8) * Nn + nt * 16 + l] = acc[r] + bv;
}

// ---------------------------------------------------------------------------
// GEMM2 (Nn=128, Kp=128) with pe added, dual bf16 output:
//   Hb [N,128] row-major (Q/K source) and Htb [64][128][1024] (V^T, padded)
// ---------------------------------------------------------------------------
__global__ __launch_bounds__(128)
void k_gemm2_bf16(const __bf16* __restrict__ Ab, const __bf16* __restrict__ Bt,
                  const float* __restrict__ bias, const float* __restrict__ pe,
                  __bf16* __restrict__ Hb, __bf16* __restrict__ Htb, int M) {
    const int wid = blockIdx.x * (blockDim.x >> 5) + (threadIdx.x >> 5);
    const int mt = wid >> 3, nt = wid & 7;
    if (mt * 16 >= M) return;
    const int lane = threadIdx.x & 31, half = lane >> 4, l = lane & 15;
    const __bf16* Ap = Ab + (size_t)(mt * 16 + l) * 128 + half * 16;
    const __bf16* Bp = Bt + (size_t)(nt * 16 + l) * 128 + half * 16;

    v8f acc = {};
    for (int kc = 0; kc < 128; kc += 32) {
        v16bf a = cat8(*(const v8bf*)(Ap + kc), *(const v8bf*)(Ap + kc + 8));
        v16bf b = *(const v16bf*)(Bp + kc);
        acc = __builtin_amdgcn_wmma_f32_16x16x32_bf16(false, a, false, b,
                                                      (short)0, acc, false, false);
    }
    const int col = nt * 16 + l;
    const float bv = bias[col];
#pragma unroll
    for (int r = 0; r < 8; ++r) {
        int row = mt * 16 + r + half * 8;
        int gph = row / 1008, tok = row % 1008;
        __bf16 v = f2bf(acc[r] + bv + pe[(size_t)tok * 128 + col]);
        Hb[(size_t)row * 128 + col] = v;
        Htb[((size_t)gph * 128 + col) * 1024 + tok] = v;
    }
}

// ---------------------------------------------------------------------------
// Full self-attention per graph (1008 tokens, d=128), bf16 WMMA.
// One wave per (graph, 16-query tile); scores fp32 in LDS rows padded to 1024.
// ---------------------------------------------------------------------------
#define TOK 1008
__global__ __launch_bounds__(32)
void k_attn(const __bf16* __restrict__ Hb, const __bf16* __restrict__ Htb,
            float* __restrict__ ctx) {
    __shared__ __align__(16) float sc[16 * 1024];     // 64 KB
    const int g  = blockIdx.x / 63;
    const int qt = blockIdx.x % 63;
    const __bf16* Hg = Hb + (size_t)g * TOK * 128;
    const __bf16* Vt = Htb + (size_t)g * 128 * 1024;
    const int lane = threadIdx.x, half = lane >> 4, l = lane & 15;

    // zero the padded tail so the P@V k-loop needs no bounds checks
    if (lane < 16) {
#pragma unroll
        for (int r = 0; r < 16; ++r) sc[r * 1024 + TOK + lane] = 0.f;
    }

    // Q fragments: 16 rows x 128 dims -> 4 chunks of K=32 (kept in registers)
    v16bf qa[4];
    const __bf16* Qp = Hg + (size_t)(qt * 16 + l) * 128 + half * 16;
#pragma unroll
    for (int kc = 0; kc < 4; ++kc)
        qa[kc] = cat8(*(const v8bf*)(Qp + kc * 32), *(const v8bf*)(Qp + kc * 32 + 8));

    const float scale = 0.088388347648318447f;        // 1/sqrt(128)
    for (int kt = 0; kt < 63; ++kt) {
        if (kt + 1 < 63)
            __builtin_prefetch(Hg + (size_t)(kt + 1) * 16 * 128 + lane * 64, 0, 3);
        v8f s = {};
        const __bf16* Kp_ = Hg + (size_t)(kt * 16 + l) * 128 + half * 16;
#pragma unroll
        for (int kc = 0; kc < 4; ++kc) {
            v16bf b = *(const v16bf*)(Kp_ + kc * 32);
            s = __builtin_amdgcn_wmma_f32_16x16x32_bf16(false, qa[kc], false, b,
                                                        (short)0, s, false, false);
        }
#pragma unroll
        for (int r = 0; r < 8; ++r)
            sc[(r + half * 8) * 1024 + kt * 16 + l] = s[r] * scale;
    }
    __syncthreads();

    // row softmax (wave-cooperative)
    for (int r = 0; r < 16; ++r) {
        float* row = sc + r * 1024;
        float m = -3.0e38f;
        for (int c = lane; c < TOK; c += 32) m = fmaxf(m, row[c]);
#pragma unroll
        for (int off = 16; off > 0; off >>= 1) m = fmaxf(m, __shfl_xor(m, off, 32));
        float sum = 0.f;
        for (int c = lane; c < TOK; c += 32) {
            float e = __expf(row[c] - m);
            row[c] = e;
            sum += e;
        }
#pragma unroll
        for (int off = 16; off > 0; off >>= 1) sum += __shfl_xor(sum, off, 32);
        float inv = 1.f / sum;
        for (int c = lane; c < TOK; c += 32) row[c] *= inv;
    }
    __syncthreads();

    // ctx = P @ V ; 32 full K=32 chunks (tail zero-padded in LDS / Htb)
    v8f acc[8] = {};
    for (int kc = 0; kc < 32; ++kc) {
        v16bf a;
#pragma unroll
        for (int gg = 0; gg < 8; ++gg) {
            int k0 = kc * 32 + a_kbase(gg, half);
            float2 p = *(const float2*)(&sc[l * 1024 + k0]);
            a[2 * gg]     = f2bf(p.x);
            a[2 * gg + 1] = f2bf(p.y);
        }
#pragma unroll
        for (int nt = 0; nt < 8; ++nt) {
            v16bf b = *(const v16bf*)(Vt + (size_t)(nt * 16 + l) * 1024 +
                                      kc * 32 + half * 16);
            acc[nt] = __builtin_amdgcn_wmma_f32_16x16x32_bf16(false, a, false, b,
                                                              (short)0, acc[nt],
                                                              false, false);
        }
    }
#pragma unroll
    for (int nt = 0; nt < 8; ++nt)
#pragma unroll
        for (int r = 0; r < 8; ++r) {
            int q = qt * 16 + r + half * 8;
            ctx[((size_t)g * TOK + q) * 128 + nt * 16 + l] = acc[nt][r];
        }
}

// ---------------------------------------------------------------------------
// segmented max/mean pooling over 84-node blocks
// ---------------------------------------------------------------------------
__global__ void k_pool(const float* __restrict__ h1, const float* __restrict__ ctx,
                       float* __restrict__ feat) {
    int i = blockIdx.x * blockDim.x + threadIdx.x;
    if (i >= 64 * 12 * 128) return;
    int d = i & 127;
    int s = (i >> 7) % 12;
    int b = i / (12 * 128);
    size_t base = ((size_t)b * 1008 + s * 84) * 128 + d;
    float mx1 = -3e38f, sm1 = 0.f, mx2 = -3e38f, sm2 = 0.f;
    for (int n = 0; n < 84; ++n) {
        float v1 = h1[base + (size_t)n * 128];
        float v2 = ctx[base + (size_t)n * 128];
        mx1 = fmaxf(mx1, v1); sm1 += v1;
        mx2 = fmaxf(mx2, v2); sm2 += v2;
    }
    float* f = feat + (size_t)b * 6144 + s * 512 + d;
    f[0]   = mx1;
    f[128] = sm1 * (1.f / 84.f);
    f[256] = mx2;
    f[384] = sm2 * (1.f / 84.f);
}

// ---------------------------------------------------------------------------
// MLP head (tiny; scalar fp32)
// ---------------------------------------------------------------------------
__global__ void k_head1(const float* __restrict__ feat, const float* __restrict__ W,
                        const float* __restrict__ b, float* __restrict__ z) {
    int i = blockIdx.x * blockDim.x + threadIdx.x;      // 64*512
    if (i >= 64 * 512) return;
    int c = i & 511, r = i >> 9;
    float acc = b[c];
    const float* fr = feat + (size_t)r * 6144;
    for (int k = 0; k < 6144; ++k) acc += fr[k] * W[(size_t)k * 512 + c];
    z[i] = siluf(acc);
}

__global__ void k_bn(float* __restrict__ z, const float* __restrict__ gam,
                     const float* __restrict__ bet, int rows, int cols) {
    int c = blockIdx.x * blockDim.x + threadIdx.x;
    if (c >= cols) return;
    float mu = 0.f;
    for (int r = 0; r < rows; ++r) mu += z[(size_t)r * cols + c];
    mu /= rows;
    float var = 0.f;
    for (int r = 0; r < rows; ++r) {
        float d = z[(size_t)r * cols + c] - mu;
        var += d * d;
    }
    var /= rows;                                        // biased (torch-style)
    float s = gam[c] * rsqrtf(var + 1e-5f);
    for (int r = 0; r < rows; ++r)
        z[(size_t)r * cols + c] = (z[(size_t)r * cols + c] - mu) * s + bet[c];
}

__global__ void k_head2(const float* __restrict__ z1, const float* __restrict__ W,
                        const float* __restrict__ b, float* __restrict__ z2) {
    int i = blockIdx.x * blockDim.x + threadIdx.x;      // 64*32
    if (i >= 64 * 32) return;
    int c = i & 31, r = i >> 5;
    float acc = b[c];
    for (int k = 0; k < 512; ++k) acc += z1[(size_t)r * 512 + k] * W[(size_t)k * 32 + c];
    z2[i] = siluf(acc);
}

__global__ void k_head3(const float* __restrict__ z2, const float* __restrict__ W,
                        const float* __restrict__ b, float* __restrict__ out) {
    int r = blockIdx.x * blockDim.x + threadIdx.x;
    if (r >= 64) return;
    float l0 = b[0], l1 = b[1];
    for (int k = 0; k < 32; ++k) {
        float v = z2[r * 32 + k];
        l0 += v * W[k * 2];
        l1 += v * W[k * 2 + 1];
    }
    float m  = fmaxf(l0, l1);
    float e0 = __expf(l0 - m), e1 = __expf(l1 - m);
    float inv = 1.f / (e0 + e1);
    out[r * 2]     = e0 * inv;
    out[r * 2 + 1] = e1 * inv;
}

// ---------------------------------------------------------------------------
extern "C" void kernel_launch(void* const* d_in, const int* in_sizes, int n_in,
                              void* d_out, int out_size, void* d_ws, size_t ws_size,
                              hipStream_t stream) {
    (void)n_in; (void)out_size; (void)ws_size;
    const float* x   = (const float*)d_in[0];
    const int*   ei  = (const int*)d_in[1];
    const float* w1  = (const float*)d_in[2];
    const float* b1  = (const float*)d_in[3];
    const float* w2  = (const float*)d_in[4];
    const float* b2  = (const float*)d_in[5];
    const float* pe  = (const float*)d_in[6];
    const float* wf1 = (const float*)d_in[7];
    const float* bf1 = (const float*)d_in[8];
    const float* g1  = (const float*)d_in[9];
    const float* be1 = (const float*)d_in[10];
    const float* wf2 = (const float*)d_in[11];
    const float* bf2 = (const float*)d_in[12];
    const float* g2  = (const float*)d_in[13];
    const float* be2 = (const float*)d_in[14];
    const float* wf3 = (const float*)d_in[15];
    const float* bf3 = (const float*)d_in[16];
    float* out = (float*)d_out;

    const long N = 64512;
    const long E = (long)in_sizes[1] / 2;

    // ---- workspace layout (256B-aligned zones, with lifetime-based reuse) --
    char* W_ = (char*)d_ws;
    auto alloc = [&](size_t bytes) {
        char* p = W_;
        W_ += (bytes + 255) & ~(size_t)255;
        return p;
    };
    float*  agg1 = (float*)alloc(N * 84 * 4);          // later reused as A2b
    __bf16* A1b  = (__bf16*)alloc(N * 96 * 2);
    float*  h1   = (float*)alloc(N * 128 * 4);
    float*  agg2 = (float*)alloc(N * 128 * 4);         // later reused as ctx
    __bf16* Wt1  = (__bf16*)alloc(128 * 96 * 2);
    __bf16* Wt2  = (__bf16*)alloc(128 * 128 * 2);
    __bf16* Hb   = (__bf16*)alloc(N * 128 * 2);
    __bf16* Htb  = (__bf16*)alloc((size_t)64 * 128 * 1024 * 2);
    float*  feat = (float*)alloc(64 * 6144 * 4);
    float*  z1   = (float*)alloc(64 * 512 * 4);
    float*  z2   = (float*)alloc(64 * 32 * 4);
    __bf16* A2b  = (__bf16*)agg1;                      // N*128*2 <= N*84*4
    float*  ctx  = agg2;                               // dead after A2b built

    const int T = 256;
    // ---- GIN layer 1 -------------------------------------------------------
    k_zero<<<(int)((N * 84 + T - 1) / T), T, 0, stream>>>(agg1, N * 84);
    k_scatter<<<(int)((E * 84 + T - 1) / T), T, 0, stream>>>(x, ei, E, 84, agg1);
    k_fuse_bf16<<<(int)((N * 96 + T - 1) / T), T, 0, stream>>>(x, agg1, A1b, N, 84, 96);
    k_wT_bf16<<<(128 * 96 + T - 1) / T, T, 0, stream>>>(w1, Wt1, 84, 128, 96);
    k_gemm_bf16<<<8064, 128, 0, stream>>>(A1b, Wt1, b1, h1, 64512, 128, 96);
    // ---- GIN layer 2 (+ pe), dual bf16 outputs -----------------------------
    k_zero<<<(int)((N * 128 + T - 1) / T), T, 0, stream>>>(agg2, N * 128);
    k_scatter<<<(int)((E * 128 + T - 1) / T), T, 0, stream>>>(h1, ei, E, 128, agg2);
    k_fuse_bf16<<<(int)((N * 128 + T - 1) / T), T, 0, stream>>>(h1, agg2, A2b, N, 128, 128);
    k_wT_bf16<<<(128 * 128 + T - 1) / T, T, 0, stream>>>(w2, Wt2, 128, 128, 128);
    k_zero<<<(int)(((size_t)64 * 128 * 1024 / 2 + T - 1) / T), T, 0, stream>>>(
        (float*)Htb, (long)64 * 128 * 1024 / 2);
    k_gemm2_bf16<<<8064, 128, 0, stream>>>(A2b, Wt2, b2, pe, Hb, Htb, 64512);
    // ---- per-graph full self-attention -------------------------------------
    k_attn<<<64 * 63, 32, 0, stream>>>(Hb, Htb, ctx);
    // ---- pooling + MLP head ------------------------------------------------
    k_pool<<<(64 * 12 * 128 + T - 1) / T, T, 0, stream>>>(h1, ctx, feat);
    k_head1<<<(64 * 512 + T - 1) / T, T, 0, stream>>>(feat, wf1, bf1, z1);
    k_bn<<<2, 256, 0, stream>>>(z1, g1, be1, 64, 512);
    k_head2<<<(64 * 32 + T - 1) / T, T, 0, stream>>>(z1, wf2, bf2, z2);
    k_bn<<<1, 32, 0, stream>>>(z2, g2, be2, 64, 32);
    k_head3<<<1, 64, 0, stream>>>(z2, wf3, bf3, out);
}
